// Attention_58274116272611
// MI455X (gfx1250) — compile-verified, matrix operand order checked by reading, not run on hardware
//
#include <hip/hip_runtime.h>
#include <hip/hip_bf16.h>

// Fused single-head attention for B=16, S=1024, H=1024 on gfx1250 (wave32, WMMA).
// Pipeline: cast f32->f16  ->  QKV projection GEMM (v_wmma_f32_16x16x32_f16)
//           ->  fused scores/softmax/attn*V kernel with 64KB LDS score tile.
// K-loops are manually unrolled x2 with explicit A/B fragment ping-pong
// (static register renaming, no v_mov copies) and sched_group_barrier hints
// interleave next-k VMEM loads into the WMMA chain (partial s_wait_loadcnt).

#define B_ 16
#define S_ 1024
#define H_ 1024

typedef __attribute__((ext_vector_type(16))) _Float16 v16h;
typedef __attribute__((ext_vector_type(8)))  _Float16 v8h;
typedef __attribute__((ext_vector_type(8)))  float    v8f;
typedef __attribute__((ext_vector_type(4)))  float    v4f;

#if __has_builtin(__builtin_amdgcn_sched_group_barrier)
#define SGB(mask, size, id) __builtin_amdgcn_sched_group_barrier((mask), (size), (id))
#else
#define SGB(mask, size, id)
#endif
#define SG_MFMA 0x008   // matrix ops (MFMA/WMMA)
#define SG_VMEMR 0x020  // VMEM reads
#define SG_DSR 0x100    // DS reads

union FragU { v16h v; v8h h[2]; };

// Load a 16x32 f16 WMMA fragment from a K-contiguous (row-major, ld elements/row)
// tile. Works for both A (row = M) and B (row = N, i.e. "NT" operand).
// ISA layout (16-bit A 16x32): lanes 0-15 hold K = k0..k0+7 (elems 0-7) and
// k0+16..k0+23 (elems 8-15); lanes 16-31 the same +8.  -> two 16B loads.
__device__ __forceinline__ v16h load_frag_nt(const _Float16* __restrict__ base,
                                             int ld, int k0, int r, int hi) {
  FragU u;
  const _Float16* p = base + (size_t)r * ld + (k0 + (hi << 3));
  u.h[0] = *(const v8h*)(p);        // K = k0+8*hi   .. +7
  u.h[1] = *(const v8h*)(p + 16);   // K = k0+8*hi+16.. +7
  return u.v;
}

__device__ __forceinline__ v8f wmma_f16(v16h a, v16h b, v8f c) {
  return __builtin_amdgcn_wmma_f32_16x16x32_f16(false, a, false, b, (short)0, c,
                                                false, false);
}

// 16-row x 128-col strip GEMM over K (multiple of 64), A and B K-contiguous
// row-major. Manually unrolled x2 with A+B fragment ping-pong.
__device__ __forceinline__ void strip_16x128(const _Float16* __restrict__ A,
                                             const _Float16* __restrict__ Bt,
                                             int ld, int K, int r, int hi,
                                             v8f acc[8]) {
  v16h aC, aN, bC[8], bN[8];
  aC = load_frag_nt(A, ld, 0, r, hi);
#pragma unroll
  for (int t = 0; t < 8; ++t)
    bC[t] = load_frag_nt(Bt + (size_t)(t * 16) * ld, ld, 0, r, hi);

  for (int k0 = 0; k0 < K; k0 += 64) {
    // ---- half 1: compute k0 with aC/bC, load k0+32 into aN/bN ----
    aN = load_frag_nt(A, ld, k0 + 32, r, hi);
#pragma unroll
    for (int t = 0; t < 8; ++t)
      bN[t] = load_frag_nt(Bt + (size_t)(t * 16) * ld, ld, k0 + 32, r, hi);
    SGB(SG_VMEMR, 2, 0);
#pragma unroll
    for (int t = 0; t < 8; ++t) {
      acc[t] = wmma_f16(aC, bC[t], acc[t]);
      SGB(SG_MFMA, 1, 0);
      SGB(SG_VMEMR, 2, 0);
    }
    // ---- half 2: compute k0+32 with aN/bN, load k0+64 into aC/bC ----
    if (k0 + 64 < K) {
      aC = load_frag_nt(A, ld, k0 + 64, r, hi);
#pragma unroll
      for (int t = 0; t < 8; ++t)
        bC[t] = load_frag_nt(Bt + (size_t)(t * 16) * ld, ld, k0 + 64, r, hi);
      __builtin_prefetch(A + (size_t)r * ld + (k0 + 128), 0, 3);
    }
    SGB(SG_VMEMR, 2, 0);
#pragma unroll
    for (int t = 0; t < 8; ++t) {
      acc[t] = wmma_f16(aN, bN[t], acc[t]);
      SGB(SG_MFMA, 1, 0);
      SGB(SG_VMEMR, 2, 0);
    }
  }
}

// ---------------------------------------------------------------------------
// Kernel 1: cast input X [B*S*H] and Wq|Wk|Wv [3*H*H] from f32 to f16.
// ---------------------------------------------------------------------------
__global__ __launch_bounds__(256) void cast_inputs_kernel(
    const float* __restrict__ x,  const float* __restrict__ Wq,
    const float* __restrict__ Wk, const float* __restrict__ Wv,
    _Float16* __restrict__ Xh, _Float16* __restrict__ Wh /*3*H*H contiguous*/) {
  size_t i = (size_t)blockIdx.x * blockDim.x + threadIdx.x;
  const size_t NX = (size_t)B_ * S_ * H_;
  const size_t HH = (size_t)H_ * H_;
  if (i < NX) { Xh[i] = (_Float16)x[i]; return; }
  size_t j = i - NX;
  if (j >= 3 * HH) return;
  int which = (int)(j / HH);
  size_t idx = j - (size_t)which * HH;
  const float* W = (which == 0) ? Wq : ((which == 1) ? Wk : Wv);
  Wh[j] = (_Float16)W[idx];
}

// ---------------------------------------------------------------------------
// Kernel 2: Q/K/V projection. out[m,n] = sum_k X[m,k]*W[n,k] + bias[n].
// grid = (M/16, 1, 3); block = 256 (8 waves). Wave w owns n in [128w,128w+128).
// Q,K stored [b][s][h] f16; V stored transposed [b][h][s] f16 for the AV GEMM.
// ---------------------------------------------------------------------------
__global__ __launch_bounds__(256) void qkv_gemm_kernel(
    const _Float16* __restrict__ Xh, const _Float16* __restrict__ Wh,
    const float* __restrict__ bq, const float* __restrict__ bk,
    const float* __restrict__ bv,
    _Float16* __restrict__ Qh, _Float16* __restrict__ Kh,
    _Float16* __restrict__ Vth) {
  const int tid  = threadIdx.x;
  const int wave = tid >> 5, lane = tid & 31;
  const int r = lane & 15, hi = lane >> 4;
  const int m0   = blockIdx.x << 4;
  const int proj = blockIdx.z;          // 0=Q, 1=K, 2=V
  const int n0   = wave << 7;

  const _Float16* W  = Wh + (size_t)proj * H_ * H_ + (size_t)n0 * H_;
  const float* bias  = (proj == 0) ? bq : ((proj == 1) ? bk : bv);
  const _Float16* Ar = Xh + (size_t)m0 * H_;

  v8f acc[8] = {};
  strip_16x128(Ar, W, H_, H_, r, hi, acc);

#pragma unroll
  for (int t = 0; t < 8; ++t) {
    const int col = n0 + t * 16 + r;
    const float bb = bias[col];
#pragma unroll
    for (int j = 0; j < 8; ++j) {
      const int row = m0 + j + (hi << 3);
      const float val = acc[t][j] + bb;
      if (proj == 0) {
        Qh[(size_t)row * H_ + col] = (_Float16)val;
      } else if (proj == 1) {
        Kh[(size_t)row * H_ + col] = (_Float16)val;
      } else {
        const int bIdx = row >> 10, sIdx = row & (S_ - 1);
        Vth[((size_t)bIdx * H_ + col) * S_ + sIdx] = (_Float16)val;
      }
    }
  }
}

// ---------------------------------------------------------------------------
// Kernel 3: fused scores + mask + softmax + attn*V for one (batch, 16-row
// query tile). 256 threads / 8 waves; 16x1024 f32 score tile in LDS (64 KB).
// ---------------------------------------------------------------------------
__global__ __launch_bounds__(256) void attn_kernel(
    const _Float16* __restrict__ Qh, const _Float16* __restrict__ Kh,
    const _Float16* __restrict__ Vth, const int* __restrict__ mask,
    float* __restrict__ outA, float* __restrict__ outW) {
  __shared__ float sc[16 * S_];       // 64 KB score / weight tile

  const int tid  = threadIdx.x;
  const int wave = tid >> 5, lane = tid & 31;
  const int r = lane & 15, hi = lane >> 4;
  const int b  = blockIdx.y;
  const int m0 = blockIdx.x << 4;
  const int n0 = wave << 7;

  // ---- Phase 1: scores = (Q . K^T) / 32, masked, into LDS ----
  const _Float16* Qrows = Qh + ((size_t)b * S_ + m0) * H_;
  const _Float16* Krows = Kh + (size_t)b * S_ * H_ + (size_t)n0 * H_;

  v8f acc[8] = {};
  strip_16x128(Qrows, Krows, H_, H_, r, hi, acc);

  const float scale = 0.03125f;  // 1/sqrt(1024)
#pragma unroll
  for (int t = 0; t < 8; ++t) {
    const int col = n0 + t * 16 + r;
    const int mv  = mask[b * S_ + col];
#pragma unroll
    for (int j = 0; j < 8; ++j) {
      const int row = j + (hi << 3);
      sc[row * S_ + col] = mv ? acc[t][j] * scale : -1e9f;
    }
  }
  __syncthreads();

  // ---- Phase 2: row softmax (2 rows per wave), write weights out ----
#pragma unroll
  for (int rr = 0; rr < 2; ++rr) {
    const int row = wave * 2 + rr;
    float* prow = sc + row * S_;
    float mx = -3.402823466e38f;
    for (int c = lane; c < S_; c += 32) mx = fmaxf(mx, prow[c]);
#pragma unroll
    for (int off = 16; off > 0; off >>= 1) mx = fmaxf(mx, __shfl_xor(mx, off, 32));
    float sum = 0.0f;
    for (int c = lane; c < S_; c += 32) {
      float e = __expf(prow[c] - mx);
      prow[c] = e;
      sum += e;
    }
#pragma unroll
    for (int off = 16; off > 0; off >>= 1) sum += __shfl_xor(sum, off, 32);
    const float inv = 1.0f / sum;
    float* gw = outW + ((size_t)b * S_ + m0 + row) * S_;
    for (int c = lane; c < S_; c += 32) {
      float w = prow[c] * inv;
      prow[c] = w;
      gw[c]   = w;
    }
  }
  __syncthreads();

  // ---- Phase 3: attended = weights(16xS) @ V, using V^T [H][S] ----
  const _Float16* Vb = Vth + (size_t)b * H_ * S_ + (size_t)n0 * S_;
  v8f acc2[8] = {};
  {
    v16h bC[8], bN[8];
#pragma unroll
    for (int t = 0; t < 8; ++t)
      bC[t] = load_frag_nt(Vb + (size_t)(t * 16) * S_, S_, 0, r, hi);

    for (int k0 = 0; k0 < S_; k0 += 64) {
      // ---- half 1: A(k0) from LDS f32 weights; load V frags for k0+32 ----
      const float* prow0 = sc + r * S_ + k0 + (hi << 3);
      v4f f0 = *(const v4f*)(prow0);
      v4f f1 = *(const v4f*)(prow0 + 4);
      v4f f2 = *(const v4f*)(prow0 + 16);
      v4f f3 = *(const v4f*)(prow0 + 20);
      v16h a0;
#pragma unroll
      for (int j = 0; j < 4; ++j) {
        a0[j]      = (_Float16)f0[j];
        a0[4 + j]  = (_Float16)f1[j];
        a0[8 + j]  = (_Float16)f2[j];
        a0[12 + j] = (_Float16)f3[j];
      }
#pragma unroll
      for (int t = 0; t < 8; ++t)
        bN[t] = load_frag_nt(Vb + (size_t)(t * 16) * S_, S_, k0 + 32, r, hi);
      SGB(SG_DSR, 4, 0);
#pragma unroll
      for (int t = 0; t < 8; ++t) {
        acc2[t] = wmma_f16(a0, bC[t], acc2[t]);
        SGB(SG_MFMA, 1, 0);
        SGB(SG_VMEMR, 2, 0);
      }
      // ---- half 2: A(k0+32); load V frags for k0+64 ----
      const float* prow1 = prow0 + 32;
      f0 = *(const v4f*)(prow1);
      f1 = *(const v4f*)(prow1 + 4);
      f2 = *(const v4f*)(prow1 + 16);
      f3 = *(const v4f*)(prow1 + 20);
      v16h a1;
#pragma unroll
      for (int j = 0; j < 4; ++j) {
        a1[j]      = (_Float16)f0[j];
        a1[4 + j]  = (_Float16)f1[j];
        a1[8 + j]  = (_Float16)f2[j];
        a1[12 + j] = (_Float16)f3[j];
      }
      if (k0 + 64 < S_) {
#pragma unroll
        for (int t = 0; t < 8; ++t)
          bC[t] = load_frag_nt(Vb + (size_t)(t * 16) * S_, S_, k0 + 64, r, hi);
      }
      SGB(SG_DSR, 4, 0);
#pragma unroll
      for (int t = 0; t < 8; ++t) {
        acc2[t] = wmma_f16(a1, bN[t], acc2[t]);
        SGB(SG_MFMA, 1, 0);
        SGB(SG_VMEMR, 2, 0);
      }
    }
  }
#pragma unroll
  for (int t = 0; t < 8; ++t) {
    const int col = n0 + t * 16 + r;
#pragma unroll
    for (int j = 0; j < 8; ++j) {
      const int row = m0 + j + (hi << 3);
      outA[((size_t)b * S_ + row) * H_ + col] = acc2[t][j];
    }
  }
}

// ---------------------------------------------------------------------------
extern "C" void kernel_launch(void* const* d_in, const int* in_sizes, int n_in,
                              void* d_out, int out_size, void* d_ws,
                              size_t ws_size, hipStream_t stream) {
  const float* x  = (const float*)d_in[0];
  const int*   mk = (const int*)d_in[1];
  const float* Wq = (const float*)d_in[2];
  const float* bq = (const float*)d_in[3];
  const float* Wk = (const float*)d_in[4];
  const float* bk = (const float*)d_in[5];
  const float* Wv = (const float*)d_in[6];
  const float* bv = (const float*)d_in[7];

  const size_t NX = (size_t)B_ * S_ * H_;   // 16,777,216
  const size_t HH = (size_t)H_ * H_;        // 1,048,576

  _Float16* ws  = (_Float16*)d_ws;
  _Float16* Xh  = ws;                       // NX halfs
  _Float16* Wh  = Xh + NX;                  // 3*HH halfs (Wq|Wk|Wv)
  _Float16* Qh  = Wh + 3 * HH;              // NX halfs
  _Float16* Kh  = Qh + NX;                  // NX halfs
  _Float16* Vth = Kh + NX;                  // NX halfs (V transposed [b][h][s])

  float* outA = (float*)d_out;              // [B,S,H] attended
  float* outW = outA + NX;                  // [B,S,S] attention weights

  const size_t ncast = NX + 3 * HH;
  cast_inputs_kernel<<<dim3((unsigned)((ncast + 255) / 256)), 256, 0, stream>>>(
      x, Wq, Wk, Wv, Xh, Wh);

  qkv_gemm_kernel<<<dim3(B_ * S_ / 16, 1, 3), 256, 0, stream>>>(
      Xh, Wh, bq, bk, bv, Qh, Kh, Vth);

  attn_kernel<<<dim3(S_ / 16, B_), 256, 0, stream>>>(
      Qh, Kh, Vth, mk, outA, outW);
}
